// RotaryTransformerEncoderLayer_58755152609846
// MI455X (gfx1250) — matrix operand order repaired and checked by reading, not compiled, not run on hardware
//
#include <hip/hip_runtime.h>
#include <hip/hip_bf16.h>

#define DEV __device__ __forceinline__

typedef __bf16 bf16;
typedef __attribute__((ext_vector_type(16))) __bf16 v16bf;
typedef __attribute__((ext_vector_type(8)))  __bf16 bf16x8;
typedef __attribute__((ext_vector_type(8)))  float  v8f;

union V16U { v16bf v; bf16x8 h[2]; };

DEV v8f wmma_bf16(v16bf a, v16bf b, v8f c) {
  return __builtin_amdgcn_wmma_f32_16x16x32_bf16(false, a, false, b, (short)0, c, false, false);
}

// async copy 16B global -> LDS, tracked by ASYNCcnt (CDNA5 ISA)
#define ASYNC_CP_B128(lds_u32, gptr)                                          \
  asm volatile("global_load_async_to_lds_b128 %0, %1, off"                    \
               :: "v"(lds_u32), "v"(gptr) : "memory")
#define WAIT_ASYNC(n) asm volatile("s_wait_asynccnt " #n ::: "memory")

constexpr int BATCH = 4, SEQ = 2048, DM = 1024, NH = 16, HD = 64, DFF = 4096;
constexpr int TOK = BATCH * SEQ; // 8192

// ---------------------------------------------------------------- cast f32->bf16
__global__ void k_cast_bf16(const float* __restrict__ in, bf16* __restrict__ out, int n) {
  int i = blockIdx.x * blockDim.x + threadIdx.x;
  int stride = gridDim.x * blockDim.x;
  for (; i < n; i += stride) out[i] = (bf16)in[i];
}

// ---------------------------------------------------------------- GEMM  C = A[M,K] * W[N,K]^T + bias
// 128x128 block tile, BK=32, double-buffered LDS filled by async-to-LDS copies.
enum { EPI_F32 = 0, EPI_RELU_BF16 = 1, EPI_BHSD = 2, EPI_VT = 3 };

template <int MODE>
__global__ __launch_bounds__(256) void k_gemm(const bf16* __restrict__ A,
                                              const bf16* __restrict__ W,
                                              const float* __restrict__ bias,
                                              void* __restrict__ out,
                                              int M, int N, int K) {
  __shared__ __align__(16) bf16 As[2][128][40];   // 128 x 32 tiles, padded rows
  __shared__ __align__(16) bf16 Bs[2][128][40];
  const int tid  = threadIdx.x;
  const int lane = tid & 31, wid = tid >> 5;
  const int wm = wid & 3, wn = wid >> 2;          // waves: 4 (M) x 2 (N), 32x64 each
  const int m0 = blockIdx.y * 128, n0 = blockIdx.x * 128;
  const int lh = lane >> 4;                       // lane-half (0/1)
  const int ll = lane & 15;

  auto fill = [&](int st, int k0) {
#pragma unroll
    for (int i = 0; i < 2; ++i) {                 // A tile: 128 rows x 4 x 16B chunks
      int c = tid + i * 256;
      int row = c >> 2, sub = c & 3;
      unsigned l = (unsigned)(uintptr_t)&As[st][row][sub * 8];
      const bf16* g = A + (size_t)(m0 + row) * K + k0 + sub * 8;
      ASYNC_CP_B128(l, g);
    }
#pragma unroll
    for (int i = 0; i < 2; ++i) {                 // B tile: 128 rows of W
      int c = tid + i * 256;
      int row = c >> 2, sub = c & 3;
      unsigned l = (unsigned)(uintptr_t)&Bs[st][row][sub * 8];
      const bf16* g = W + (size_t)(n0 + row) * K + k0 + sub * 8;
      ASYNC_CP_B128(l, g);
    }
  };

  v8f acc[2][4] = {};
  const int nsteps = K >> 5;
  fill(0, 0);
  for (int s = 0; s < nsteps; ++s) {
    const int cur = s & 1;
    if (s + 1 < nsteps) {
      fill(cur ^ 1, (s + 1) << 5);   // prefetch next stage (overlaps wait + compute)
      WAIT_ASYNC(0x4);               // in-order: stage s (4 older ops) complete
    } else {
      WAIT_ASYNC(0x0);
    }
    __syncthreads();

    V16U a[2], bfr[4];
#pragma unroll
    for (int mi = 0; mi < 2; ++mi) {
      // A 16x32 bf16 fragment: lanes<16 rows M, halves K[0..7] & K[16..23]; lanes>=16 K[8..15] & K[24..31]
      const bf16* ap = &As[cur][wm * 32 + mi * 16 + ll][lh * 8];
      a[mi].h[0] = *(const bf16x8*)ap;
      a[mi].h[1] = *(const bf16x8*)(ap + 16);
    }
#pragma unroll
    for (int ni = 0; ni < 4; ++ni) {
      // B 32x16 bf16 fragment: lanes<16 col N, K[0..7] & K[16..23]; lanes>=16 K[8..15] & K[24..31]
      const bf16* bp = &Bs[cur][wn * 64 + ni * 16 + ll][lh * 8];
      bfr[ni].h[0] = *(const bf16x8*)bp;
      bfr[ni].h[1] = *(const bf16x8*)(bp + 16);
    }
#pragma unroll
    for (int mi = 0; mi < 2; ++mi)
#pragma unroll
      for (int ni = 0; ni < 4; ++ni)
        acc[mi][ni] = wmma_bf16(a[mi].v, bfr[ni].v, acc[mi][ni]);
    __syncthreads();
  }

#pragma unroll
  for (int mi = 0; mi < 2; ++mi)
#pragma unroll
    for (int ni = 0; ni < 4; ++ni)
#pragma unroll
      for (int r = 0; r < 8; ++r) {
        int gm = m0 + wm * 32 + mi * 16 + lh * 8 + r;   // D: VGPR r -> row r (half0) / 8+r (half1)
        int gn = n0 + wn * 64 + ni * 16 + ll;
        float v = acc[mi][ni][r] + bias[gn];
        if constexpr (MODE == EPI_F32) {
          ((float*)out)[(size_t)gm * N + gn] = v;
        } else if constexpr (MODE == EPI_RELU_BF16) {
          ((bf16*)out)[(size_t)gm * N + gn] = (bf16)fmaxf(v, 0.f);
        } else if constexpr (MODE == EPI_BHSD) {
          int b = gm / SEQ, s = gm - b * SEQ;
          int h = gn >> 6, d = gn & 63;
          ((bf16*)out)[((size_t)(b * NH + h) * SEQ + s) * HD + d] = (bf16)v;
        } else { // EPI_VT: V stored [B,H,HD,S] so attention B-frags are contiguous
          int b = gm / SEQ, s = gm - b * SEQ;
          int h = gn >> 6, d = gn & 63;
          ((bf16*)out)[((size_t)(b * NH + h) * HD + d) * SEQ + s] = (bf16)v;
        }
      }
}

// ---------------------------------------------------------------- RoPE in-place on [BH,S,HD] bf16
__global__ void k_rope(bf16* __restrict__ qk) {
  int idx = blockIdx.x * blockDim.x + threadIdx.x;     // over BH*S*32
  if (idx >= BATCH * NH * SEQ * 32) return;
  int d  = idx & 31;
  int s  = (idx >> 5) & (SEQ - 1);
  int bh = idx >> 16;
  bf16* p = qk + ((size_t)bh * SEQ + s) * HD;
  float inv = __expf(-(float)d * 0.28782313662425572f);  // ln(10000)/32
  float sn, cs;
  __sincosf((float)s * inv, &sn, &cs);
  float x1 = (float)p[d], x2 = (float)p[d + 32];
  p[d]      = (bf16)(x1 * cs - x2 * sn);
  p[d + 32] = (bf16)(x2 * cs + x1 * sn);
}

// ---------------------------------------------------------------- flash attention (non-causal)
__global__ __launch_bounds__(128) void k_flash(const bf16* __restrict__ q,
                                               const bf16* __restrict__ k,
                                               const bf16* __restrict__ vt,
                                               bf16* __restrict__ attn) {
  __shared__ __align__(16) bf16 P[4][16][40];   // per-wave 16x32 P tile
  const int lane = threadIdx.x & 31, wid = threadIdx.x >> 5;
  const int lh = lane >> 4, ll = lane & 15;
  const int bh = blockIdx.y;                    // b*NH + h
  const int b = bh >> 4, h = bh & 15;
  const int q0 = blockIdx.x * 64 + wid * 16;
  const size_t head = (size_t)bh * SEQ * HD;

  V16U aq[2];  // Q 16x64 as two 16x32 A-fragments
  {
    const bf16* qp = q + head + (size_t)(q0 + ll) * HD;
#pragma unroll
    for (int kk = 0; kk < 2; ++kk) {
      const bf16* p = qp + kk * 32 + lh * 8;
      aq[kk].h[0] = *(const bf16x8*)p;
      aq[kk].h[1] = *(const bf16x8*)(p + 16);
    }
  }

  float mrun[8], lrun[8];
  v8f accO[4] = {};
#pragma unroll
  for (int r = 0; r < 8; ++r) { mrun[r] = -1e30f; lrun[r] = 0.f; }
  const float scale = 0.125f;  // 1/sqrt(64)

  for (int t0 = 0; t0 < SEQ; t0 += 32) {
    v8f sc[2];
#pragma unroll
    for (int nt = 0; nt < 2; ++nt) {   // two 16-key tiles
      const bf16* kp = k + head + (size_t)(t0 + nt * 16 + ll) * HD;
      V16U b0, b1;
      b0.h[0] = *(const bf16x8*)(kp + lh * 8);
      b0.h[1] = *(const bf16x8*)(kp + lh * 8 + 16);
      b1.h[0] = *(const bf16x8*)(kp + 32 + lh * 8);
      b1.h[1] = *(const bf16x8*)(kp + 32 + lh * 8 + 16);
      v8f s = {};
      s = wmma_bf16(aq[0].v, b0.v, s);
      s = wmma_bf16(aq[1].v, b1.v, s);
      sc[nt] = s;
    }
    float p0[8], p1[8];
#pragma unroll
    for (int r = 0; r < 8; ++r) {
      float s0 = sc[0][r] * scale, s1 = sc[1][r] * scale;
      float tm = fmaxf(s0, s1);
      tm = fmaxf(tm, __shfl_xor(tm, 1));
      tm = fmaxf(tm, __shfl_xor(tm, 2));
      tm = fmaxf(tm, __shfl_xor(tm, 4));
      tm = fmaxf(tm, __shfl_xor(tm, 8));       // row max within 16-lane half
      float mn = fmaxf(mrun[r], tm);
      float al = __expf(mrun[r] - mn);
      mrun[r] = mn;
      p0[r] = __expf(s0 - mn);
      p1[r] = __expf(s1 - mn);
      float rs = p0[r] + p1[r];
      rs += __shfl_xor(rs, 1);
      rs += __shfl_xor(rs, 2);
      rs += __shfl_xor(rs, 4);
      rs += __shfl_xor(rs, 8);
      lrun[r] = lrun[r] * al + rs;
#pragma unroll
      for (int nt = 0; nt < 4; ++nt) accO[nt][r] *= al;
    }
    // D-layout P -> LDS -> A-layout fragment
#pragma unroll
    for (int r = 0; r < 8; ++r) {
      int row = lh * 8 + r;
      P[wid][row][ll]      = (bf16)p0[r];
      P[wid][row][ll + 16] = (bf16)p1[r];
    }
    asm volatile("s_wait_dscnt 0" ::: "memory");  // in-wave DS order; block compiler reordering
    V16U ap;
    {
      const bf16* pp = &P[wid][ll][lh * 8];
      ap.h[0] = *(const bf16x8*)pp;
      ap.h[1] = *(const bf16x8*)(pp + 16);
    }
    const bf16* vbase = vt + (size_t)bh * HD * SEQ + t0 + lh * 8;
#pragma unroll
    for (int nt = 0; nt < 4; ++nt) {
      V16U bv;
      const bf16* vp = vbase + (size_t)(nt * 16 + ll) * SEQ;
      bv.h[0] = *(const bf16x8*)vp;
      bv.h[1] = *(const bf16x8*)(vp + 16);
      accO[nt] = wmma_bf16(ap.v, bv.v, accO[nt]);
    }
  }
#pragma unroll
  for (int nt = 0; nt < 4; ++nt)
#pragma unroll
    for (int r = 0; r < 8; ++r) {
      int row = q0 + lh * 8 + r;
      int dim = nt * 16 + ll;
      attn[(size_t)(b * SEQ + row) * DM + h * HD + dim] = (bf16)(accO[nt][r] / lrun[r]);
    }
}

// ---------------------------------------------------------------- residual + LayerNorm
template <bool WRITE_BF>
__global__ __launch_bounds__(256) void k_addln(const float* a, const float* res,
                                               const float* __restrict__ g,
                                               const float* __restrict__ beta,
                                               float* outf, bf16* outb) {
  __shared__ float red[256];
  const int row = blockIdx.x;
  const int tid = threadIdx.x;
  float x[4];
  float s = 0.f, sq = 0.f;
#pragma unroll
  for (int i = 0; i < 4; ++i) {
    int c = tid * 4 + i;
    x[i] = a[(size_t)row * DM + c] + res[(size_t)row * DM + c];
    s += x[i]; sq += x[i] * x[i];
  }
  red[tid] = s;
  __syncthreads();
  for (int o = 128; o > 0; o >>= 1) { if (tid < o) red[tid] += red[tid + o]; __syncthreads(); }
  float mu = red[0] * (1.f / DM);
  __syncthreads();
  red[tid] = sq;
  __syncthreads();
  for (int o = 128; o > 0; o >>= 1) { if (tid < o) red[tid] += red[tid + o]; __syncthreads(); }
  float var = red[0] * (1.f / DM) - mu * mu;
  float rstd = rsqrtf(var + 1e-5f);
#pragma unroll
  for (int i = 0; i < 4; ++i) {
    int c = tid * 4 + i;
    float y = (x[i] - mu) * rstd * g[c] + beta[c];
    outf[(size_t)row * DM + c] = y;
    if constexpr (WRITE_BF) outb[(size_t)row * DM + c] = (bf16)y;
  }
}

// ---------------------------------------------------------------- launch
extern "C" void kernel_launch(void* const* d_in, const int* in_sizes, int n_in,
                              void* d_out, int out_size, void* d_ws, size_t ws_size,
                              hipStream_t stream) {
  const float* src = (const float*)d_in[0];
  const float* Wq  = (const float*)d_in[1];
  const float* bq  = (const float*)d_in[2];
  const float* Wk  = (const float*)d_in[3];
  const float* bk  = (const float*)d_in[4];
  const float* Wv  = (const float*)d_in[5];
  const float* bv  = (const float*)d_in[6];
  const float* Wo  = (const float*)d_in[7];
  const float* bo  = (const float*)d_in[8];
  const float* W1  = (const float*)d_in[9];
  const float* b1  = (const float*)d_in[10];
  const float* W2  = (const float*)d_in[11];
  const float* b2  = (const float*)d_in[12];
  const float* g1  = (const float*)d_in[13];
  const float* be1 = (const float*)d_in[14];
  const float* g2  = (const float*)d_in[15];
  const float* be2 = (const float*)d_in[16];
  float* out = (float*)d_out;

  char* ws = (char*)d_ws;
  size_t off = 0;
  auto take = [&](size_t bytes) { char* p = ws + off; off = (off + bytes + 255) & ~(size_t)255; return p; };
  bf16*  xbf    = (bf16*) take((size_t)TOK * DM * 2);
  bf16*  wqb    = (bf16*) take((size_t)DM * DM * 2);
  bf16*  wkb    = (bf16*) take((size_t)DM * DM * 2);
  bf16*  wvb    = (bf16*) take((size_t)DM * DM * 2);
  bf16*  wob    = (bf16*) take((size_t)DM * DM * 2);
  bf16*  w1b    = (bf16*) take((size_t)DFF * DM * 2);
  bf16*  w2b    = (bf16*) take((size_t)DM * DFF * 2);
  bf16*  q_r    = (bf16*) take((size_t)TOK * DM * 2);   // [B,H,S,64]
  bf16*  k_r    = (bf16*) take((size_t)TOK * DM * 2);   // [B,H,S,64]
  bf16*  v_t    = (bf16*) take((size_t)TOK * DM * 2);   // [B,H,64,S]
  bf16*  attnb  = (bf16*) take((size_t)TOK * DM * 2);   // [TOK, DM]
  float* attnof = (float*)take((size_t)TOK * DM * 4);
  bf16*  x1bf   = (bf16*) take((size_t)TOK * DM * 2);
  bf16*  hbf    = (bf16*) take((size_t)TOK * DFF * 2);
  float* fff    = (float*)take((size_t)TOK * DM * 4);
  float* x1f    = out;                                   // reuse d_out as f32 residual buffer

  // casts
  k_cast_bf16<<<dim3(1024), dim3(256), 0, stream>>>(src, xbf, TOK * DM);
  k_cast_bf16<<<dim3(512),  dim3(256), 0, stream>>>(Wq, wqb, DM * DM);
  k_cast_bf16<<<dim3(512),  dim3(256), 0, stream>>>(Wk, wkb, DM * DM);
  k_cast_bf16<<<dim3(512),  dim3(256), 0, stream>>>(Wv, wvb, DM * DM);
  k_cast_bf16<<<dim3(512),  dim3(256), 0, stream>>>(Wo, wob, DM * DM);
  k_cast_bf16<<<dim3(1024), dim3(256), 0, stream>>>(W1, w1b, DFF * DM);
  k_cast_bf16<<<dim3(1024), dim3(256), 0, stream>>>(W2, w2b, DM * DFF);

  // Q/K/V projections (WMMA + async-to-LDS), scatter into attention-friendly layouts
  dim3 gP(DM / 128, TOK / 128);
  k_gemm<EPI_BHSD><<<gP, 256, 0, stream>>>(xbf, wqb, bq, q_r, TOK, DM, DM);
  k_gemm<EPI_BHSD><<<gP, 256, 0, stream>>>(xbf, wkb, bk, k_r, TOK, DM, DM);
  k_gemm<EPI_VT>  <<<gP, 256, 0, stream>>>(xbf, wvb, bv, v_t, TOK, DM, DM);

  // RoPE in place
  int nrope = BATCH * NH * SEQ * 32;
  k_rope<<<dim3(nrope / 256), 256, 0, stream>>>(q_r);
  k_rope<<<dim3(nrope / 256), 256, 0, stream>>>(k_r);

  // flash attention
  k_flash<<<dim3(SEQ / 64, BATCH * NH), 128, 0, stream>>>(q_r, k_r, v_t, attnb);

  // output projection
  k_gemm<EPI_F32><<<gP, 256, 0, stream>>>(attnb, wob, bo, attnof, TOK, DM, DM);

  // residual + LN 1  -> x1 (f32 in d_out, bf16 copy for FFN)
  k_addln<true><<<dim3(TOK), 256, 0, stream>>>(attnof, src, g1, be1, x1f, x1bf);

  // FFN
  k_gemm<EPI_RELU_BF16><<<dim3(DFF / 128, TOK / 128), 256, 0, stream>>>(x1bf, w1b, b1, hbf, TOK, DFF, DM);
  k_gemm<EPI_F32>      <<<dim3(DM / 128,  TOK / 128), 256, 0, stream>>>(hbf, w2b, b2, fff, TOK, DM, DFF);

  // residual + LN 2 -> d_out (reads x1f==d_out before writing, per-thread private)
  k_addln<false><<<dim3(TOK), 256, 0, stream>>>(fff, x1f, g2, be2, out, nullptr);

  (void)in_sizes; (void)n_in; (void)out_size; (void)ws_size;
}
